// MultiHeadAttention_3839700762945
// MI455X (gfx1250) — matrix operand order repaired and checked by reading, not compiled, not run on hardware
//
#include <hip/hip_runtime.h>
#include <hip/hip_bf16.h>

typedef _Float16 v16h __attribute__((ext_vector_type(16)));
typedef _Float16 h8   __attribute__((ext_vector_type(8)));
typedef float    v8f  __attribute__((ext_vector_type(8)));
typedef float    fv4  __attribute__((ext_vector_type(4)));

union AF { v16h v; h8 h[2]; };

#define DIM_   1024
#define S_LEN  2048
#define HD     64
#define MROWS  8192   // B*S

#define USE_ASYNC_LDS 1

// ---- async global->LDS copy of 16 bytes (per lane), CDNA5 path ----
static __device__ inline void cp16(_Float16* lds, const _Float16* g) {
#if USE_ASYNC_LDS
    unsigned off = (unsigned)(size_t)lds;   // low 32 bits of generic shared ptr = LDS byte addr
    asm volatile("global_load_async_to_lds_b128 %0, %1, off"
                 :: "v"(off), "v"(g) : "memory");
#else
    *(h8*)lds = *(const h8*)g;
#endif
}
static __device__ inline void async_wait() {
#if USE_ASYNC_LDS
    asm volatile("s_wait_asynccnt 0x0" ::: "memory");
#endif
}

static __device__ inline v8f wmma_f16(const AF& a, const AF& b, v8f c) {
    return __builtin_amdgcn_wmma_f32_16x16x32_f16(false, a.v, false, b.v,
                                                  (short)0, c, false, false);
}

// ---- max-reduce across the 16 lanes sharing a C-layout row (VALU permlane) ----
static __device__ inline float rowmax16(float x) {
#if __has_builtin(__builtin_amdgcn_permlane16)
    unsigned u, t;
    u = __builtin_bit_cast(unsigned, x);
    t = __builtin_amdgcn_permlane16(u, u, 0x67452301u, 0xEFCDAB89u, false, false); // xor 1
    x = fmaxf(x, __builtin_bit_cast(float, t)); u = __builtin_bit_cast(unsigned, x);
    t = __builtin_amdgcn_permlane16(u, u, 0x54761032u, 0xDCFE98BAu, false, false); // xor 2
    x = fmaxf(x, __builtin_bit_cast(float, t)); u = __builtin_bit_cast(unsigned, x);
    t = __builtin_amdgcn_permlane16(u, u, 0x32107654u, 0xBA98FEDCu, false, false); // xor 4
    x = fmaxf(x, __builtin_bit_cast(float, t)); u = __builtin_bit_cast(unsigned, x);
    t = __builtin_amdgcn_permlane16(u, u, 0xFEDCBA98u, 0x76543210u, false, false); // xor 8
    x = fmaxf(x, __builtin_bit_cast(float, t));
#else
    x = fmaxf(x, __shfl_xor(x, 1, 32));
    x = fmaxf(x, __shfl_xor(x, 2, 32));
    x = fmaxf(x, __shfl_xor(x, 4, 32));
    x = fmaxf(x, __shfl_xor(x, 8, 32));
#endif
    return x;
}

// ---- 16-element f32 -> f16 convert via v_cvt_pk_rtz_f16_f32 ----
static __device__ inline void cvt16(const float* __restrict__ s, _Float16* d) {
    const fv4* s4 = (const fv4*)s;
    union { h8 h; unsigned u[4]; } r0, r1;
#pragma unroll
    for (int j = 0; j < 2; ++j) {
        fv4 v = s4[j];
        r0.u[2*j]   = __builtin_bit_cast(unsigned, __builtin_amdgcn_cvt_pkrtz(v[0], v[1]));
        r0.u[2*j+1] = __builtin_bit_cast(unsigned, __builtin_amdgcn_cvt_pkrtz(v[2], v[3]));
        fv4 w = s4[2 + j];
        r1.u[2*j]   = __builtin_bit_cast(unsigned, __builtin_amdgcn_cvt_pkrtz(w[0], w[1]));
        r1.u[2*j+1] = __builtin_bit_cast(unsigned, __builtin_amdgcn_cvt_pkrtz(w[2], w[3]));
    }
    *(h8*)&d[0] = r0.h;
    *(h8*)&d[8] = r1.h;
}

// ---- weight transpose: W[k][n] f32 -> WT[n][k] f16 ----
__global__ __launch_bounds__(256) void w_transpose_f16(const float* __restrict__ W,
                                                       _Float16* __restrict__ WT) {
    int idx = blockIdx.x * 256 + threadIdx.x;   // over 1024*1024
    int k = idx >> 10, n = idx & 1023;
    WT[(size_t)n * DIM_ + k] = (_Float16)W[idx];
}

// ---- tiled GEMM: X[M,K] @ WT[N,K]^T + bias, double-buffered async LDS ----
// MODE 0: out f16 head-split [B,H,S,hd];  MODE 1: out f32 row-major [M,N];
// MODE 2: out f16 d-major   [B,H,hd,S]
template <typename InT, int MODE>
__global__ __launch_bounds__(256) void gemm128(const InT* __restrict__ X,
                                               const _Float16* __restrict__ WT,
                                               const float* __restrict__ bias,
                                               void* __restrict__ outp,
                                               float outScale,
                                               int M, int N, int K) {
    __shared__ _Float16 As[2][128 * 32];   // 2 x 8 KB
    __shared__ _Float16 Bs[2][128 * 32];   // 2 x 8 KB

    const int tid  = threadIdx.x;
    const int wave = tid >> 5, lane = tid & 31;
    const int li = lane & 15, hi = lane >> 4;
    const int kh = hi * 8;              // A-frag K base
    const int bh16 = hi * 16;           // unused; B-frags also use kh
    const int m0 = blockIdx.y * 128, n0 = blockIdx.x * 128;
    const int wm = wave >> 1, wn = wave & 1;   // 4x2 wave grid -> 32x64 per wave
    const int ldrow = tid >> 1, ldseg = (tid & 1) * 16;

    v8f acc[2][4] = {};

    auto stage = [&](int buf, int k0) {
        const _Float16* srcB = WT + (size_t)(n0 + ldrow) * K + k0 + ldseg;
        cp16(&Bs[buf][ldrow * 32 + ldseg],     srcB);
        cp16(&Bs[buf][ldrow * 32 + ldseg + 8], srcB + 8);
        if (sizeof(InT) == 2) {         // f16 input: async copy
            const _Float16* srcH =
                (const _Float16*)(const void*)(X + (size_t)(m0 + ldrow) * K + k0 + ldseg);
            cp16(&As[buf][ldrow * 32 + ldseg],     srcH);
            cp16(&As[buf][ldrow * 32 + ldseg + 8], srcH + 8);
        } else {                        // f32 input: load + pack-convert
            const InT* src = X + (size_t)(m0 + ldrow) * K + k0 + ldseg;
            _Float16 tmp[16];
            cvt16((const float*)src, tmp);
            *(h8*)&As[buf][ldrow * 32 + ldseg]     = *(h8*)&tmp[0];
            *(h8*)&As[buf][ldrow * 32 + ldseg + 8] = *(h8*)&tmp[8];
        }
    };

    stage(0, 0);
    async_wait();
    __syncthreads();

    for (int k0 = 0; k0 < K; k0 += 32) {
        const int cur = (k0 >> 5) & 1, nxt = cur ^ 1;
        if (k0 + 32 < K) stage(nxt, k0 + 32);   // overlap DMA with WMMAs below

        AF a[2], b[4];
#pragma unroll
        for (int mt = 0; mt < 2; ++mt) {
            int m = wm * 32 + mt * 16 + li;
            a[mt].h[0] = *(const h8*)&As[cur][m * 32 + kh];
            a[mt].h[1] = *(const h8*)&As[cur][m * 32 + 16 + kh];
        }
#pragma unroll
        for (int nt = 0; nt < 4; ++nt) {
            int n = wn * 64 + nt * 16 + li;
            b[nt].h[0] = *(const h8*)&Bs[cur][n * 32 + kh];
            b[nt].h[1] = *(const h8*)&Bs[cur][n * 32 + 16 + kh];
        }
#pragma unroll
        for (int mt = 0; mt < 2; ++mt)
#pragma unroll
            for (int nt = 0; nt < 4; ++nt)
                acc[mt][nt] = wmma_f16(a[mt], b[nt], acc[mt][nt]);

        async_wait();
        __syncthreads();
    }

    // epilogue: C/D layout -> lane holds (m = r + 8*hi, n = li)
#pragma unroll
    for (int mt = 0; mt < 2; ++mt) {
#pragma unroll
        for (int nt = 0; nt < 4; ++nt) {
            int n = n0 + wn * 64 + nt * 16 + li;
            float bv = bias[n];
#pragma unroll
            for (int r = 0; r < 8; ++r) {
                int m = m0 + wm * 32 + mt * 16 + 8 * hi + r;
                float val = (acc[mt][nt][r] + bv) * outScale;
                int sIdx = m & (S_LEN - 1), bI = m >> 11;
                int hI = n >> 6, d = n & 63;
                if (MODE == 0) {
                    ((_Float16*)outp)[(((size_t)bI * 16 + hI) * S_LEN + sIdx) * HD + d] =
                        (_Float16)val;
                } else if (MODE == 2) {
                    ((_Float16*)outp)[(((size_t)bI * 16 + hI) * HD + d) * S_LEN + sIdx] =
                        (_Float16)val;
                } else {
                    ((float*)outp)[(size_t)m * N + n] = val;
                }
            }
        }
    }
}

// ---- flash attention: per block (b,h, 128 q rows); 8 waves x 16 q rows ----
// qp,kp: [B,H,S,hd] f16 (q pre-scaled by log2e/sqrt(hd) -> exp2-domain softmax);
// vpT: [B,H,hd,S] f16 (d-major)
__global__ __launch_bounds__(256) void attn_kernel(const _Float16* __restrict__ qp,
                                                   const _Float16* __restrict__ kp,
                                                   const _Float16* __restrict__ vpT,
                                                   _Float16* __restrict__ attnOut) {
    __shared__ _Float16 Ks[2][32 * HD];     // [key][d]   2 x 4 KB
    __shared__ _Float16 VTs[2][HD * 32];    // [d][key]   2 x 4 KB
    __shared__ _Float16 Ps[8][16 * 32];     // per-wave P 8 KB

    const int tid  = threadIdx.x;
    const int wave = tid >> 5, lane = tid & 31;
    const int li = lane & 15, hi = lane >> 4;
    const int kh = hi * 8, bh16 = hi * 16;
    const int bh = blockIdx.y;              // b*16 + h
    const int qb = blockIdx.x * 128;
    const int bI = bh >> 4, hI = bh & 15;

    // Q A-fragments; wave owns rows qb+wave*16..+15
    const _Float16* qbase = qp + ((size_t)bh * S_LEN + qb + wave * 16) * HD;
    AF qa[2];
    qa[0].h[0] = *(const h8*)&qbase[(size_t)li * HD + kh];
    qa[0].h[1] = *(const h8*)&qbase[(size_t)li * HD + 16 + kh];
    qa[1].h[0] = *(const h8*)&qbase[(size_t)li * HD + 32 + kh];
    qa[1].h[1] = *(const h8*)&qbase[(size_t)li * HD + 48 + kh];

    AF ones;                                // B-frag of 1.0 for row sums via WMMA
#pragma unroll
    for (int i = 0; i < 16; ++i) ones.v[i] = (_Float16)1.0f;

    float mrow[8], lrow[8];
    v8f o[4] = {};
#pragma unroll
    for (int r = 0; r < 8; ++r) { mrow[r] = -3.0e38f; lrow[r] = 0.0f; }

    const int ldkey = tid >> 3, ldd  = (tid & 7) * 8;   // K-stage coords (32x64)
    const int vd    = tid >> 2, vseg = (tid & 3) * 8;   // V-stage coords (64x32)

    auto stage = [&](int buf, int j0) {
        cp16(&Ks[buf][ldkey * HD + ldd], &kp[((size_t)bh * S_LEN + j0 + ldkey) * HD + ldd]);
        cp16(&VTs[buf][vd * 32 + vseg],  &vpT[((size_t)bh * HD + vd) * S_LEN + j0 + vseg]);
    };

    stage(0, 0);
    async_wait();
    __syncthreads();

    for (int j0 = 0; j0 < S_LEN; j0 += 32) {
        const int cur = (j0 >> 5) & 1, nxt = cur ^ 1;
        if (j0 + 32 < S_LEN) stage(nxt, j0 + 32);   // overlap DMA with compute

        // scores (exp2 domain): tile s_g holds keys 2*li+g in column li
        v8f s0 = {}, s1 = {};
        AF kb;
        kb.h[0] = *(const h8*)&Ks[cur][(2 * li + 0) * HD + kh];
        kb.h[1] = *(const h8*)&Ks[cur][(2 * li + 0) * HD + 16 + kh];
        s0 = wmma_f16(qa[0], kb, s0);
        kb.h[0] = *(const h8*)&Ks[cur][(2 * li + 0) * HD + 32 + kh];
        kb.h[1] = *(const h8*)&Ks[cur][(2 * li + 0) * HD + 48 + kh];
        s0 = wmma_f16(qa[1], kb, s0);
        kb.h[0] = *(const h8*)&Ks[cur][(2 * li + 1) * HD + kh];
        kb.h[1] = *(const h8*)&Ks[cur][(2 * li + 1) * HD + 16 + kh];
        s1 = wmma_f16(qa[0], kb, s1);
        kb.h[0] = *(const h8*)&Ks[cur][(2 * li + 1) * HD + 32 + kh];
        kb.h[1] = *(const h8*)&Ks[cur][(2 * li + 1) * HD + 48 + kh];
        s1 = wmma_f16(qa[1], kb, s1);

        // online softmax, base-2 (row m = r + 8*hi); sums via ones-WMMA below
#pragma unroll
        for (int r = 0; r < 8; ++r) {
            float a = s0[r], b = s1[r];
            float mx = rowmax16(fmaxf(a, b));
            float mnew = fmaxf(mrow[r], mx);
            float p0 = __builtin_exp2f(a - mnew), p1 = __builtin_exp2f(b - mnew);
            float corr = __builtin_exp2f(mrow[r] - mnew);
            mrow[r] = mnew;
            lrow[r] *= corr;
#pragma unroll
            for (int j = 0; j < 4; ++j) o[j][r] *= corr;
            // packed f16x2 store: keys 2li, 2li+1 are adjacent columns
            *(unsigned*)&Ps[wave][(r + 8 * hi) * 32 + 2 * li] =
                __builtin_bit_cast(unsigned, __builtin_amdgcn_cvt_pkrtz(p0, p1));
        }
        __threadfence_block();   // order per-wave LDS P writes before A-frag reads

        AF pa, vb;
        pa.h[0] = *(const h8*)&Ps[wave][li * 32 + kh];
        pa.h[1] = *(const h8*)&Ps[wave][li * 32 + 16 + kh];

        // row sums of P in one WMMA: P(16x32) x ones(32x16) -> rowsum in every col
        v8f srow = {};
        srow = wmma_f16(pa, ones, srow);
#pragma unroll
        for (int r = 0; r < 8; ++r) lrow[r] += srow[r];

        // O += P * V  (V^T rows as B-fragments)
#pragma unroll
        for (int nt = 0; nt < 4; ++nt) {
            int d0 = nt * 16 + li;
            vb.h[0] = *(const h8*)&VTs[cur][d0 * 32 + kh];
            vb.h[1] = *(const h8*)&VTs[cur][d0 * 32 + 16 + kh];
            o[nt] = wmma_f16(pa, vb, o[nt]);
        }

        async_wait();
        __syncthreads();
    }

    // normalize + store to attnOut[B*S, DIM] (col = h*64 + d), f16
#pragma unroll
    for (int r = 0; r < 8; ++r) {
        float inv = 1.0f / lrow[r];
        int s = qb + wave * 16 + 8 * hi + r;
        size_t rowbase = ((size_t)bI * S_LEN + s) * DIM_ + hI * HD;
#pragma unroll
        for (int nt = 0; nt < 4; ++nt)
            attnOut[rowbase + nt * 16 + li] = (_Float16)(o[nt][r] * inv);
    }
}

extern "C" void kernel_launch(void* const* d_in, const int* in_sizes, int n_in,
                              void* d_out, int out_size, void* d_ws, size_t ws_size,
                              hipStream_t stream) {
    const float* Q   = (const float*)d_in[0];
    const float* Kin = (const float*)d_in[1];
    const float* V   = (const float*)d_in[2];
    const float* Wq  = (const float*)d_in[3];
    const float* bq  = (const float*)d_in[4];
    const float* Wk  = (const float*)d_in[5];
    const float* bk  = (const float*)d_in[6];
    const float* Wv  = (const float*)d_in[7];
    const float* bv  = (const float*)d_in[8];
    const float* Wo  = (const float*)d_in[9];
    const float* bo  = (const float*)d_in[10];
    float* out = (float*)d_out;

    char* ws = (char*)d_ws;
    const size_t MB = 1024 * 1024;
    _Float16* WqT     = (_Float16*)(ws + 0 * MB);    // 2 MB each
    _Float16* WkT     = (_Float16*)(ws + 2 * MB);
    _Float16* WvT     = (_Float16*)(ws + 4 * MB);
    _Float16* WoT     = (_Float16*)(ws + 6 * MB);
    _Float16* qp      = (_Float16*)(ws + 8 * MB);    // 16 MB each
    _Float16* kp      = (_Float16*)(ws + 24 * MB);
    _Float16* vpT     = (_Float16*)(ws + 40 * MB);   // d-major [B,H,hd,S]
    _Float16* attnOut = (_Float16*)(ws + 56 * MB);   // 16 MB, [B*S, DIM]

    dim3 tb(256);

    w_transpose_f16<<<4096, tb, 0, stream>>>(Wq, WqT);
    w_transpose_f16<<<4096, tb, 0, stream>>>(Wk, WkT);
    w_transpose_f16<<<4096, tb, 0, stream>>>(Wv, WvT);
    w_transpose_f16<<<4096, tb, 0, stream>>>(Wo, WoT);

    dim3 gg(DIM_ / 128, MROWS / 128);   // (8, 64)
    // fold 1/sqrt(HEAD_DIM) * log2(e) into q so the softmax runs in exp2 domain
    const float qscale = 0.125f * 1.44269504088896340736f;
    gemm128<float, 0><<<gg, tb, 0, stream>>>(Q,   WqT, bq, qp,  qscale, MROWS, DIM_, DIM_);
    gemm128<float, 0><<<gg, tb, 0, stream>>>(Kin, WkT, bk, kp,  1.0f,   MROWS, DIM_, DIM_);
    gemm128<float, 2><<<gg, tb, 0, stream>>>(V,   WvT, bv, vpT, 1.0f,   MROWS, DIM_, DIM_);

    dim3 ga(S_LEN / 128, 4 * 16);       // (16 q-blocks, B*H)
    attn_kernel<<<ga, tb, 0, stream>>>(qp, kp, vpT, attnOut);

    gemm128<_Float16, 1><<<gg, tb, 0, stream>>>(attnOut, WoT, bo, out, 1.0f, MROWS, DIM_, DIM_);
}
